// GATLayer_21457656611238
// MI455X (gfx1250) — matrix operand order repaired and checked by reading, not compiled, not run on hardware
//
#include <hip/hip_runtime.h>

typedef __attribute__((ext_vector_type(16))) __bf16          v16bf;
typedef __attribute__((ext_vector_type(8)))  float           v8f;
typedef unsigned short ushort_t;
typedef __attribute__((ext_vector_type(8)))  ushort_t        v8us;
typedef __attribute__((ext_vector_type(16))) ushort_t        v16us;

#define GAT_M   64
#define GAT_IN  64
#define GAT_HID 64
#define XPAD    72   // bf16 row stride: 144 B, multiple of 16 B for b128 LDS loads
#define TPAD    68   // f32 row stride: 272 B, multiple of 16 B for float4 stores

// round-to-nearest-even f32 -> bf16 bits
static __device__ __forceinline__ unsigned bf_bits(float x) {
    unsigned u = __float_as_uint(x);
    return (u + 0x7FFFu + ((u >> 16) & 1u)) >> 16;
}
static __device__ __forceinline__ unsigned pack_bf2(float a, float b) {
    return bf_bits(a) | (bf_bits(b) << 16);
}

// Build a v16bf WMMA fragment from two contiguous 8-element (16 B) LDS chunks.
static __device__ __forceinline__ v16bf load_frag(const ushort_t* p0, const ushort_t* p1) {
    v8us lo = *(const v8us*)p0;
    v8us hi = *(const v8us*)p1;
    v16us u = __builtin_shufflevector(lo, hi, 0,1,2,3,4,5,6,7,8,9,10,11,12,13,14,15);
    return __builtin_bit_cast(v16bf, u);
}

__global__ __launch_bounds__(128) void gat_layer_kernel(
        const float* __restrict__ item, const float* __restrict__ r,
        const float* __restrict__ t,    const float* __restrict__ mask,
        const float* __restrict__ W,    const float* __restrict__ bias,
        const float* __restrict__ hvec, float* __restrict__ out) {

    const int b   = blockIdx.x;
    const int tid = threadIdx.x;

    __shared__ __attribute__((aligned(16))) ushort_t sW[GAT_HID][XPAD]; // W bf16, row h contiguous in i
    __shared__ __attribute__((aligned(16))) ushort_t sX[GAT_M][XPAD];   // (r*t) bf16
    __shared__ __attribute__((aligned(16))) float    sT[GAT_M][TPAD];   // t fp32
    __shared__ float sBias[GAT_HID], sH[GAT_HID], sProj[GAT_M], sS[GAT_M];

    // --- Stage W (bf16), bias, h into LDS; L2-resident across blocks ---
    const float4* W4 = (const float4*)W;
    #pragma unroll
    for (int j = 0; j < 8; ++j) {
        int f = j * 128 + tid;                 // float4 index, 1024 total
        int row = f >> 4, col = (f & 15) << 2;
        float4 wv = W4[f];
        uint2 pk;
        pk.x = pack_bf2(wv.x, wv.y);
        pk.y = pack_bf2(wv.z, wv.w);
        *(uint2*)&sW[row][col] = pk;
    }
    if (tid < GAT_HID) { sBias[tid] = bias[tid]; sH[tid] = hvec[tid]; }

    // --- Stage r,t; compute x = r*t in bf16; keep t in fp32 ---
    const float4* r4 = (const float4*)(r + (size_t)b * (GAT_M * GAT_IN));
    const float4* t4 = (const float4*)(t + (size_t)b * (GAT_M * GAT_IN));
    #pragma unroll
    for (int j = 0; j < 8; ++j) {
        int f = j * 128 + tid;
        int row = f >> 4, col = (f & 15) << 2;
        float4 rv = r4[f], tv = t4[f];
        uint2 pk;
        pk.x = pack_bf2(rv.x * tv.x, rv.y * tv.y);
        pk.y = pack_bf2(rv.z * tv.z, rv.w * tv.w);
        *(uint2*)&sX[row][col] = pk;
        *(float4*)&sT[row][col] = tv;
    }
    __syncthreads();

    // --- 64x64x64 GEMM: wave w owns rows [16w,16w+16); 4 col-tiles x 2 K-chunks ---
    const int w     = tid >> 5;
    const int l     = tid & 31;
    const int lhalf = l >> 4;       // 0: lanes 0-15, 1: lanes 16-31
    const int l16   = l & 15;

    v8f acc[4];
    #pragma unroll
    for (int c = 0; c < 4; ++c)
        #pragma unroll
        for (int v = 0; v < 8; ++v) acc[c][v] = 0.0f;

    #pragma unroll
    for (int kk = 0; kk < 2; ++kk) {
        const int k0   = kk * 32;
        const int arow = 16 * w + l16;
        const int ac0  = k0 + (lhalf ? 16 : 0);          // A: 16 contiguous K per half
        v16bf afrag = load_frag(&sX[arow][ac0], &sX[arow][ac0 + 8]);
        #pragma unroll
        for (int c = 0; c < 4; ++c) {
            const int n  = 16 * c + l16;                 // B column -> W row h
            const int kb = k0 + (lhalf ? 16 : 0);        // B: 16 contiguous K per half
            v16bf bfrag = load_frag(&sW[n][kb], &sW[n][kb + 8]);
            acc[c] = __builtin_amdgcn_wmma_f32_16x16x32_bf16(
                         false, afrag, false, bfrag, (short)0, acc[c], false, false);
        }
    }

    // --- proj[m] = sum_h relu(score + bias)*h, reduced across the 16-lane half ---
    float partial[8];
    #pragma unroll
    for (int v = 0; v < 8; ++v) partial[v] = 0.0f;
    #pragma unroll
    for (int c = 0; c < 4; ++c) {
        const int N  = 16 * c + l16;
        const float bn = sBias[N], hn = sH[N];
        #pragma unroll
        for (int v = 0; v < 8; ++v) {
            float sc = acc[c][v] + bn;
            sc = sc > 0.0f ? sc : 0.0f;
            partial[v] += sc * hn;
        }
    }
    #pragma unroll
    for (int v = 0; v < 8; ++v) {
        #pragma unroll
        for (int off = 1; off < 16; off <<= 1)
            partial[v] += __shfl_xor(partial[v], off, 32);
    }
    if (l16 == 0) {
        #pragma unroll
        for (int v = 0; v < 8; ++v)
            sProj[16 * w + v + 8 * lhalf] = partial[v];
    }
    __syncthreads();

    // --- masked exp (no max-subtraction, matching reference) ---
    if (tid < GAT_M)
        sS[tid] = mask[(size_t)b * GAT_M + tid] * __expf(sProj[tid]);
    __syncthreads();

    // --- out[i] = item[i] + sum_m (s[m]/sum s) * t[m][i] ---
    if (tid < GAT_IN) {
        float tot = 0.0f;
        #pragma unroll 8
        for (int m = 0; m < GAT_M; ++m) tot += sS[m];
        float accv = 0.0f;
        #pragma unroll 8
        for (int m = 0; m < GAT_M; ++m) accv += sS[m] * sT[m][tid];
        const size_t o = (size_t)b * GAT_IN + tid;
        out[o] = item[o] + accv / tot;
    }
}

extern "C" void kernel_launch(void* const* d_in, const int* in_sizes, int n_in,
                              void* d_out, int out_size, void* d_ws, size_t ws_size,
                              hipStream_t stream) {
    (void)n_in; (void)out_size; (void)d_ws; (void)ws_size;
    const float* item = (const float*)d_in[0];   // (B, 64)
    const float* r    = (const float*)d_in[1];   // (B, 64, 64)
    const float* t    = (const float*)d_in[2];   // (B, 64, 64)
    const float* mask = (const float*)d_in[3];   // (B, 64)
    const float* W    = (const float*)d_in[4];   // (64, 64)
    const float* bias = (const float*)d_in[5];   // (64,)
    const float* h    = (const float*)d_in[6];   // (64, 1)
    float* out        = (float*)d_out;           // (B, 64)

    const int B = in_sizes[0] / GAT_IN;          // 8192
    gat_layer_kernel<<<dim3(B), dim3(128), 0, stream>>>(item, r, t, mask, W, bias, h, out);
}
